// PointCloudAE_46119358825127
// MI455X (gfx1250) — matrix operand order repaired and checked by reading, not compiled, not run on hardware
//
#include <hip/hip_runtime.h>
#include <hip/hip_bf16.h>

#define BATCH   8
#define NPTS0   8192
#define MGRID   4096
#define LATENTC 512

typedef __attribute__((ext_vector_type(16))) _Float16 v16h;
typedef __attribute__((ext_vector_type(8)))  float    v8f;

union F16x16 { uint4 u4[2]; v16h v; };

static inline int P64(int c) { return (c + 63) & ~63; }

// ---------------------------------------------------------------------------
// x (B,3,N) -> coords (B,N,3)  (f32)
// ---------------------------------------------------------------------------
__global__ void k_transpose_in(const float* __restrict__ x, float* __restrict__ coords) {
  int e = blockIdx.x * blockDim.x + threadIdx.x;
  const int total = BATCH * NPTS0 * 3;
  if (e >= total) return;
  int b = e / (NPTS0 * 3);
  int r = e % (NPTS0 * 3);
  int n = r / 3;
  int c = r % 3;
  coords[e] = x[((size_t)b * 3 + c) * NPTS0 + n];
}

// coords f32 -> f16 copy (level-0 point features)
__global__ void k_f32_to_f16(const float* __restrict__ src, _Float16* __restrict__ dst, int n) {
  int e = blockIdx.x * blockDim.x + threadIdx.x;
  if (e < n) dst[e] = (_Float16)src[e];
}

// ---------------------------------------------------------------------------
// Weight staging: W (ci,co) f32 -> Wst (coP x ciP) f16 N-major, zero padded.
// bias (co) f32 -> biasP (coP) f32 zero padded.
// ---------------------------------------------------------------------------
__global__ void k_prep_weights(const float* __restrict__ W, const float* __restrict__ b,
                               _Float16* __restrict__ Wst, float* __restrict__ biasP,
                               int ci, int co, int ciP, int coP) {
  int e = blockIdx.x * blockDim.x + threadIdx.x;
  int total = coP * ciP;
  if (e < total) {
    int n = e / ciP, k = e % ciP;
    float v = (n < co && k < ci) ? W[(size_t)k * co + n] : 0.0f;
    Wst[e] = (_Float16)v;
  }
  if (e < coP) biasP[e] = (e < co) ? b[e] : 0.0f;
}

// ---------------------------------------------------------------------------
// Farthest point sampling (unchanged; latency bound, one block per batch).
// ---------------------------------------------------------------------------
__global__ void __launch_bounds__(256) k_fps(const float* __restrict__ coords,
                                             int Npts, int S, int* __restrict__ fps_idx) {
  const int b = blockIdx.x;
  const int t = threadIdx.x;
  const int T = 256;
  const float* C = coords + (size_t)b * Npts * 3;
  float dist[32];
  const int ppt = (Npts + T - 1) / T;
  for (int i = 0; i < ppt; ++i) dist[i] = 1e10f;
  __shared__ float sd[256];
  __shared__ int   si[256];
  int far = 0;
  for (int s = 0; s < S; ++s) {
    if (t == 0) fps_idx[b * S + s] = far;
    float cx = C[far * 3 + 0], cy = C[far * 3 + 1], cz = C[far * 3 + 2];
    float best = -1.0f; int bi = 0;
    for (int i = 0; i < ppt; ++i) {
      int n = t + i * T;
      if (n < Npts) {
        float dx = C[n * 3 + 0] - cx, dy = C[n * 3 + 1] - cy, dz = C[n * 3 + 2] - cz;
        float d = dx * dx + dy * dy + dz * dz;
        float dm = fminf(dist[i], d);
        dist[i] = dm;
        if (dm > best) { best = dm; bi = n; }
      }
    }
    sd[t] = best; si[t] = bi;
    __syncthreads();
    for (int off = 128; off > 0; off >>= 1) {
      if (t < off) {
        if (sd[t + off] > sd[t]) { sd[t] = sd[t + off]; si[t] = si[t + off]; }
      }
      __syncthreads();
    }
    far = si[0];
    __syncthreads();
  }
}

__global__ void k_gather_newxyz(const float* __restrict__ coords,
                                const int* __restrict__ fps_idx,
                                int Npts, int S, float* __restrict__ nx) {
  int q = blockIdx.x * blockDim.x + threadIdx.x;
  if (q >= BATCH * S) return;
  int b = q / S;
  int idx = fps_idx[q];
  for (int c = 0; c < 3; ++c)
    nx[q * 3 + c] = coords[((size_t)b * Npts + idx) * 3 + c];
}

__global__ void k_ball_query(const float* __restrict__ coords,
                             const float* __restrict__ nx,
                             int Npts, int S, int K, float r2,
                             int* __restrict__ gidx) {
  int q = blockIdx.x * blockDim.x + threadIdx.x;
  if (q >= BATCH * S) return;
  int b = q / S;
  const float* C = coords + (size_t)b * Npts * 3;
  float qx = nx[q * 3 + 0], qy = nx[q * 3 + 1], qz = nx[q * 3 + 2];
  int* o = gidx + (size_t)q * K;
  int cnt = 0, first = -1;
  for (int n = 0; n < Npts && cnt < K; ++n) {
    float dx = C[n * 3 + 0] - qx, dy = C[n * 3 + 1] - qy, dz = C[n * 3 + 2] - qz;
    if (dx * dx + dy * dy + dz * dz <= r2) {
      if (first < 0) first = n;
      o[cnt++] = n;
    }
  }
  if (first < 0) first = Npts - 1;
  for (; cnt < K; ++cnt) o[cnt] = first;
}

// ---------------------------------------------------------------------------
// Grouped feature rows -> f16 A matrix with padded lda, zero-filled pad.
// ---------------------------------------------------------------------------
__global__ void k_gather_group(const float* __restrict__ coords,
                               const _Float16* __restrict__ points,
                               const float* __restrict__ nx,
                               const int* __restrict__ gidx,
                               int Npts, int S, int K, int Cpts, int lda,
                               _Float16* __restrict__ A) {
  int row = blockIdx.x;           // B*S*K
  int bs  = row / K;
  int b   = bs / S;
  int g   = gidx[row];
  _Float16* a = A + (size_t)row * lda;
  const _Float16* pp = points + ((size_t)b * Npts + g) * Cpts;
  const float* cc = coords + ((size_t)b * Npts + g) * 3;
  const float* qq = nx + (size_t)bs * 3;
  for (int c = threadIdx.x; c < lda; c += blockDim.x) {
    float v = 0.0f;
    if (c < Cpts) v = (float)pp[c];
    else if (c < Cpts + 3) v = cc[c - Cpts] - qq[c - Cpts];
    a[c] = (_Float16)v;
  }
}

// ---------------------------------------------------------------------------
// WMMA f16 GEMM:  D16(rows,coP) = act(A16(rows,ciP) @ Wst^T + biasP)
// All dims pre-padded (rows % 128 == 0, ciP % 64 == 0, coP % 64 == 0).
// Block tile 128x64, 8 waves, wave tile 32x32 -> 4 WMMAs / K-step.
// No bounds checks anywhere in the hot loop; pure b128 staging.
// ---------------------------------------------------------------------------
#define GBM 128
#define GBN 64
#define LSTR 40   // halves; 80B row: 16B aligned, bank-spread

__global__ void __launch_bounds__(256) k_gemm_f16(
    const _Float16* __restrict__ A16, const _Float16* __restrict__ Wst,
    const float* __restrict__ biasP, _Float16* __restrict__ D16,
    int rows, int ciP, int coP, int relu) {
  __shared__ __align__(16) _Float16 As[GBM][LSTR];
  __shared__ __align__(16) _Float16 Ws[GBN][LSTR];
  const int t     = threadIdx.x;
  const int lane  = t & 31;
  const int wv    = t >> 5;
  const int bm    = blockIdx.x * GBM;
  const int bn    = blockIdx.y * GBN;
  const int tm    = (wv & 3) * 32;   // wave row tile
  const int tn    = (wv >> 2) * 32;  // wave col tile
  const int mrow  = lane & 15;
  const int khalf = lane >> 4;

  v8f acc00 = {}, acc01 = {}, acc10 = {}, acc11 = {};

  const _Float16* Abase = A16 + (size_t)bm * ciP;
  const _Float16* Wbase = Wst + (size_t)bn * ciP;

  for (int kb = 0; kb < ciP; kb += 32) {
    // Stage A tile: 128x32 halves = 512 uint4, 2 per thread.
#pragma unroll
    for (int i = 0; i < 2; ++i) {
      int e = t + i * 256;
      int r = e >> 2, k8 = e & 3;
      uint4 d = *(const uint4*)(Abase + (size_t)r * ciP + kb + k8 * 8);
      *(uint4*)&As[r][k8 * 8] = d;
    }
    // Stage W tile: 64x32 halves = 256 uint4, 1 per thread.
    {
      int r = t >> 2, k8 = t & 3;
      uint4 d = *(const uint4*)(Wbase + (size_t)r * ciP + kb + k8 * 8);
      *(uint4*)&Ws[r][k8 * 8] = d;
    }
    __syncthreads();

    // Fragments (ISA 7.12.2): A lane m=L&15, K chunks [kh*8..+7],[16+kh*8..+7];
    // B lane n=L&15, K chunks [kh*8..+7],[16+kh*8..+7].
    F16x16 a0, a1, b0, b1;
    a0.u4[0] = *(const uint4*)&As[tm + mrow][khalf * 8];
    a0.u4[1] = *(const uint4*)&As[tm + mrow][16 + khalf * 8];
    a1.u4[0] = *(const uint4*)&As[tm + 16 + mrow][khalf * 8];
    a1.u4[1] = *(const uint4*)&As[tm + 16 + mrow][16 + khalf * 8];
    b0.u4[0] = *(const uint4*)&Ws[tn + mrow][khalf * 8];
    b0.u4[1] = *(const uint4*)&Ws[tn + mrow][16 + khalf * 8];
    b1.u4[0] = *(const uint4*)&Ws[tn + 16 + mrow][khalf * 8];
    b1.u4[1] = *(const uint4*)&Ws[tn + 16 + mrow][16 + khalf * 8];

    acc00 = __builtin_amdgcn_wmma_f32_16x16x32_f16(false, a0.v, false, b0.v, (short)0, acc00, false, false);
    acc01 = __builtin_amdgcn_wmma_f32_16x16x32_f16(false, a0.v, false, b1.v, (short)0, acc01, false, false);
    acc10 = __builtin_amdgcn_wmma_f32_16x16x32_f16(false, a1.v, false, b0.v, (short)0, acc10, false, false);
    acc11 = __builtin_amdgcn_wmma_f32_16x16x32_f16(false, a1.v, false, b1.v, (short)0, acc11, false, false);
    __syncthreads();
  }

  // Epilogue: C/D layout lane -> n = L&15, m = i + 8*(L>>4). Unguarded stores
  // (pads were computed as exact zeros via zero-padded W/bias).
  float bias0 = biasP[bn + tn + mrow];
  float bias1 = biasP[bn + tn + 16 + mrow];
  int n0 = bn + tn + mrow;
  int n1 = bn + tn + 16 + mrow;
#pragma unroll
  for (int i = 0; i < 8; ++i) {
    int m0 = bm + tm + i + 8 * khalf;
    int m1 = m0 + 16;
    float v;
    v = acc00[i] + bias0; if (relu) v = fmaxf(v, 0.0f);
    D16[(size_t)m0 * coP + n0] = (_Float16)v;
    v = acc01[i] + bias1; if (relu) v = fmaxf(v, 0.0f);
    D16[(size_t)m0 * coP + n1] = (_Float16)v;
    v = acc10[i] + bias0; if (relu) v = fmaxf(v, 0.0f);
    D16[(size_t)m1 * coP + n0] = (_Float16)v;
    v = acc11[i] + bias1; if (relu) v = fmaxf(v, 0.0f);
    D16[(size_t)m1 * coP + n1] = (_Float16)v;
  }
}

// ---------------------------------------------------------------------------
// Max-pool over K neighbors (f16 in, f16 out into concatenated points).
// ---------------------------------------------------------------------------
__global__ void k_maxpool(const _Float16* __restrict__ h, int S, int K, int C, int ld,
                          _Float16* __restrict__ outp, int Ctot, int coff) {
  int e = blockIdx.x * blockDim.x + threadIdx.x;
  if (e >= BATCH * S * C) return;
  int bs = e / C, c = e % C;
  const _Float16* hp = h + ((size_t)bs * K) * ld + c;
  float m = -3.4e38f;
  for (int k = 0; k < K; ++k) m = fmaxf(m, (float)hp[(size_t)k * ld]);
  outp[(size_t)bs * Ctot + coff + c] = (_Float16)m;
}

// z[b,c] = max over 16 points (f16 in, f32 out)
__global__ void k_zmax(const _Float16* __restrict__ pts, float* __restrict__ z) {
  int e = blockIdx.x * blockDim.x + threadIdx.x;
  if (e >= BATCH * LATENTC) return;
  int b = e / LATENTC, c = e % LATENTC;
  float m = -3.4e38f;
  for (int s = 0; s < 16; ++s) m = fmaxf(m, (float)pts[(size_t)(b * 16 + s) * LATENTC + c]);
  z[e] = m;
}

// h0[(b*M+m), :] = [ z[b] , dec_grid[m] , zero pad ]  (f16, lda padded)
__global__ void k_build_h0(const float* __restrict__ z, const float* __restrict__ grid,
                           _Float16* __restrict__ h0, int lda) {
  int row = blockIdx.x;  // B*M
  int b = row / MGRID, m = row % MGRID;
  _Float16* o = h0 + (size_t)row * lda;
  for (int c = threadIdx.x; c < lda; c += blockDim.x) {
    float v = 0.0f;
    if (c < LATENTC) v = z[b * LATENTC + c];
    else if (c < LATENTC + 3) v = grid[m * 3 + (c - LATENTC)];
    o[c] = (_Float16)v;
  }
}

// recon: (B*M, ld=64) f16 -> (B,3,M) f32
__global__ void k_write_recon(const _Float16* __restrict__ h, float* __restrict__ out, int ld) {
  int e = blockIdx.x * blockDim.x + threadIdx.x;
  if (e >= BATCH * 3 * MGRID) return;
  int b = e / (3 * MGRID);
  int r = e % (3 * MGRID);
  int c = r / MGRID;
  int m = r % MGRID;
  out[e] = (float)h[((size_t)(b * MGRID + m)) * ld + c];
}

__global__ void k_write_z(const float* __restrict__ z, float* __restrict__ out) {
  int e = blockIdx.x * blockDim.x + threadIdx.x;
  if (e >= BATCH * LATENTC) return;
  out[e] = z[e];
}

// ---------------------------------------------------------------------------
// Host orchestration
// ---------------------------------------------------------------------------
struct LayerDesc {
  const float* W; const float* b;      // raw f32 params
  int ci, co, ciP, coP;
  _Float16* Wst; float* bP;            // staged
};

static void prep_layer(LayerDesc& L, hipStream_t s) {
  int total = L.coP * L.ciP;
  k_prep_weights<<<(total + 255) / 256, 256, 0, s>>>(L.W, L.b, L.Wst, L.bP,
                                                     L.ci, L.co, L.ciP, L.coP);
}

static void run_gemm(const _Float16* A, const LayerDesc& L, _Float16* D, int rows,
                     int relu, hipStream_t s) {
  dim3 g(rows / GBM, L.coP / GBN);
  k_gemm_f16<<<g, 256, 0, s>>>(A, L.Wst, L.bP, D, rows, L.ciP, L.coP, relu);
}

extern "C" void kernel_launch(void* const* d_in, const int* in_sizes, int n_in,
                              void* d_out, int out_size, void* d_ws, size_t ws_size,
                              hipStream_t stream) {
  (void)in_sizes; (void)n_in; (void)out_size; (void)ws_size;
  const float* x = (const float*)d_in[0];

  static const int mlp[4][2][3] = {
      {{16, 16, 32},    {32, 32, 64}},
      {{64, 64, 128},   {64, 96, 128}},
      {{128, 196, 256}, {128, 196, 256}},
      {{256, 256, 256}, {256, 384, 256}}};
  static const int cin0[4]  = {6, 99, 259, 515};
  static const int S_arr[4] = {1024, 256, 64, 16};
  static const float rad[4][2] = {{0.05f, 0.1f}, {0.1f, 0.2f}, {0.2f, 0.4f}, {0.4f, 0.8f}};

  LayerDesc sa[4][2][3];
  int p = 1;
  for (int l = 0; l < 4; ++l)
    for (int sc = 0; sc < 2; ++sc)
      for (int ly = 0; ly < 3; ++ly) {
        LayerDesc& L = sa[l][sc][ly];
        L.W = (const float*)d_in[p++];
        L.b = (const float*)d_in[p++];
        L.ci = (ly == 0) ? cin0[l] : mlp[l][sc][ly - 1];
        L.co = mlp[l][sc][ly];
        L.ciP = P64(L.ci); L.coP = P64(L.co);
      }
  const float* dec_grid = (const float*)d_in[p++];
  static const int dm_dims[5] = {LATENTC + 3, 1024, 512, 256, 128};
  LayerDesc dm[4];
  for (int i = 0; i < 4; ++i) {
    dm[i].W = (const float*)d_in[p++]; dm[i].b = (const float*)d_in[p++];
    dm[i].ci = dm_dims[i]; dm[i].co = dm_dims[i + 1];
    dm[i].ciP = P64(dm[i].ci); dm[i].coP = P64(dm[i].co);
  }
  static const int dr_dims[4] = {128, 128, 64, 3};
  LayerDesc dr[3];
  for (int i = 0; i < 3; ++i) {
    dr[i].W = (const float*)d_in[p++]; dr[i].b = (const float*)d_in[p++];
    dr[i].ci = dr_dims[i]; dr[i].co = dr_dims[i + 1];
    dr[i].ciP = P64(dr[i].ci); dr[i].coP = P64(dr[i].co);
  }

  // ---- workspace carve-out ----
  char* base = (char*)d_ws;
  size_t off = 0;
  auto alloc = [&](size_t bytes) -> void* {
    void* r = (void*)(base + off);
    off += (bytes + 255) & ~(size_t)255;
    return r;
  };
  float*     coordsA = (float*)alloc(sizeof(float) * (size_t)BATCH * NPTS0 * 3);
  float*     coordsB = (float*)alloc(sizeof(float) * (size_t)BATCH * NPTS0 * 3);
  _Float16*  pts0    = (_Float16*)alloc(sizeof(_Float16) * (size_t)BATCH * NPTS0 * 3);
  _Float16*  pointsA = (_Float16*)alloc(sizeof(_Float16) * (size_t)BATCH * 1024 * 96);
  _Float16*  pointsB = (_Float16*)alloc(sizeof(_Float16) * (size_t)BATCH * 1024 * 96);
  int*       fps_idx = (int*)alloc(sizeof(int) * (size_t)BATCH * 1024);
  int*       gidx    = (int*)alloc(sizeof(int) * (size_t)BATCH * 1024 * 32);
  float*     zbuf    = (float*)alloc(sizeof(float) * (size_t)BATCH * LATENTC);
  _Float16*  bigA    = (_Float16*)alloc(sizeof(_Float16) * (size_t)BATCH * MGRID * 1024);
  _Float16*  bigB    = (_Float16*)alloc(sizeof(_Float16) * (size_t)BATCH * MGRID * 1024);

  LayerDesc* all[31]; int na = 0;
  for (int l = 0; l < 4; ++l)
    for (int sc = 0; sc < 2; ++sc)
      for (int ly = 0; ly < 3; ++ly) all[na++] = &sa[l][sc][ly];
  for (int i = 0; i < 4; ++i) all[na++] = &dm[i];
  for (int i = 0; i < 3; ++i) all[na++] = &dr[i];
  for (int i = 0; i < na; ++i) {
    all[i]->Wst = (_Float16*)alloc(sizeof(_Float16) * (size_t)all[i]->coP * all[i]->ciP);
    all[i]->bP  = (float*)alloc(sizeof(float) * (size_t)all[i]->coP);
  }
  for (int i = 0; i < na; ++i) prep_layer(*all[i], stream);

  // ---- encoder ----
  {
    int total = BATCH * NPTS0 * 3;
    k_transpose_in<<<(total + 255) / 256, 256, 0, stream>>>(x, coordsA);
    k_f32_to_f16<<<(total + 255) / 256, 256, 0, stream>>>(coordsA, pts0, total);
  }
  float*    coords_cur = coordsA;
  float*    coords_nxt = coordsB;
  _Float16* points_cur = pts0;    // level-0 points == coords (3 channels, f16)
  _Float16* points_nxt = pointsA;
  int Ncur = NPTS0, Cpts = 3;

  for (int l = 0; l < 4; ++l) {
    int S = S_arr[l];
    k_fps<<<BATCH, 256, 0, stream>>>(coords_cur, Ncur, S, fps_idx);
    int q = BATCH * S;
    k_gather_newxyz<<<(q + 127) / 128, 128, 0, stream>>>(coords_cur, fps_idx, Ncur, S, coords_nxt);
    int Ctot = mlp[l][0][2] + mlp[l][1][2];
    int coff = 0;
    for (int sc = 0; sc < 2; ++sc) {
      int K = (sc == 0) ? 16 : 32;
      float r = rad[l][sc];
      k_ball_query<<<(q + 127) / 128, 128, 0, stream>>>(coords_cur, coords_nxt, Ncur, S, K, r * r, gidx);
      int rows = BATCH * S * K;            // always a multiple of 128
      int lda0 = sa[l][sc][0].ciP;
      k_gather_group<<<rows, 128, 0, stream>>>(coords_cur, points_cur, coords_nxt,
                                               gidx, Ncur, S, K, Cpts, lda0, bigA);
      _Float16* h = bigA; _Float16* o = bigB;
      for (int ly = 0; ly < 3; ++ly) {
        run_gemm(h, sa[l][sc][ly], o, rows, 1, stream);
        _Float16* tt = h; h = o; o = tt;
      }
      int Cout = mlp[l][sc][2];
      int tot = BATCH * S * Cout;
      k_maxpool<<<(tot + 255) / 256, 256, 0, stream>>>(h, S, K, Cout, sa[l][sc][2].coP,
                                                       points_nxt, Ctot, coff);
      coff += Cout;
    }
    Ncur = S; Cpts = Ctot;
    { float* t = coords_cur; coords_cur = coords_nxt; coords_nxt = t; }
    _Float16* tmp = points_nxt;
    points_nxt = (points_nxt == pointsA) ? pointsB : pointsA;
    points_cur = tmp;
  }

  // ---- latent + decoder ----
  k_zmax<<<(BATCH * LATENTC + 255) / 256, 256, 0, stream>>>(points_cur, zbuf);
  k_build_h0<<<BATCH * MGRID, 128, 0, stream>>>(zbuf, dec_grid, bigA, dm[0].ciP);

  _Float16* h = bigA; _Float16* o = bigB;
  for (int i = 0; i < 4; ++i) {
    run_gemm(h, dm[i], o, BATCH * MGRID, (i < 3) ? 1 : 0, stream);
    _Float16* tt = h; h = o; o = tt;
  }
  for (int i = 0; i < 3; ++i) {
    run_gemm(h, dr[i], o, BATCH * MGRID, (i < 2) ? 1 : 0, stream);
    _Float16* tt = h; h = o; o = tt;
  }

  // ---- outputs ----
  float* outp = (float*)d_out;
  k_write_recon<<<(BATCH * 3 * MGRID + 255) / 256, 256, 0, stream>>>(h, outp, dr[2].coP);
  k_write_z<<<(BATCH * LATENTC + 255) / 256, 256, 0, stream>>>(zbuf, outp + (size_t)BATCH * 3 * MGRID);
}
